// Gemma4SlidingAttentionBlock_47115791237736
// MI455X (gfx1250) — matrix operand order repaired and checked by reading, not compiled, not run on hardware
//
#include <hip/hip_runtime.h>
#include <hip/hip_bf16.h>
#include <math.h>
#include <stdint.h>

// ---------------------------------------------------------------------------
// Gemma-style sliding-window attention block for MI455X (gfx1250, wave32).
// Matmuls on v_wmma_f32_16x16x32_bf16; GEMM tiles staged with CDNA5
// global_load_async_to_lds_b128 (ASYNCcnt) + double-buffered LDS.
// ---------------------------------------------------------------------------

typedef __bf16 bf16_t;
typedef __attribute__((ext_vector_type(16))) __bf16 v16bf;
typedef __attribute__((ext_vector_type(8)))  __bf16 v8bf;
typedef __attribute__((ext_vector_type(8)))  float  v8f;

#define H_SIZE   2048
#define N_HEADS  16
#define N_KV     8
#define HEAD_DIM 128
#define INTER    8192
#define WINDOW   1024
#define SEQ      2048
#define BATCH    2
#define TOKENS   (BATCH * SEQ)   // 4096

__device__ __forceinline__ v8f v8f_zero() {
  v8f z;
#pragma unroll
  for (int i = 0; i < 8; ++i) z[i] = 0.0f;
  return z;
}

__device__ __forceinline__ v8f wmma_bf16(v16bf a, v16bf b, v8f c) {
  // D = A(16x32) * B(32x16) + C(16x16), f32 accumulate
  return __builtin_amdgcn_wmma_f32_16x16x32_bf16(
      /*neg_a=*/false, a, /*neg_b=*/false, b,
      /*c_mod=*/(short)0, c, /*reuse_a=*/false, /*reuse_b=*/false);
}

// Per-lane async DMA: 16 bytes global -> LDS, tracked by ASYNCcnt.
// LDS operand = low 32 bits of the generic pointer.
__device__ __forceinline__ void async_ld_b128(void* lds, const void* g) {
  asm volatile("global_load_async_to_lds_b128 %0, %1, off"
               :: "v"((uint32_t)(uintptr_t)lds), "v"(g)
               : "memory");
}

// A-fragment (16x32, 16-bit): lane<16 holds M=lane, K={0..7,16..23};
// lane>=16 holds M=lane-16, K={8..15,24..31}.  rowp points at K=0 of row M.
__device__ __forceinline__ v16bf load_frag_a(const bf16_t* rowp, int akb) {
  union { v16bf v; v8bf h[2]; } u;
  u.h[0] = *(const v8bf*)(rowp + akb);
  u.h[1] = *(const v8bf*)(rowp + 16 + akb);
  return u.v;
}

// B-fragment (32x16, 16-bit): lane<16 holds N=lane, K={0..7,16..23};
// lane>=16 holds N=lane-16, K={8..15,24..31}.  rowp points at K=0 of column N's row.
__device__ __forceinline__ v16bf load_frag_b(const bf16_t* rowp, int bkb) {
  union { v16bf v; v8bf h[2]; } u;
  u.h[0] = *(const v8bf*)(rowp + bkb);
  u.h[1] = *(const v8bf*)(rowp + 16 + bkb);
  return u.v;
}

// ---------------------------------------------------------------------------
// GEMM: C[M,N] = A[M,K] * W[N,K]^T  (A,W bf16 row-major; C f32)
// Block tile 128x128x32, 256 threads = 8 waves; wave tile 32x64 (2x4 WMMA).
// Double-buffered LDS; tiles fetched with async-to-LDS DMA that overlaps the
// WMMAs of the previous tile. The last K-tile is peeled so the steady-state
// loop is branch-free: issue next tile -> s_wait_asynccnt 4 -> barrier ->
// 8x WMMA -> barrier.
// ---------------------------------------------------------------------------
__global__ __launch_bounds__(256)
void gemm_bf16_nt(const bf16_t* __restrict__ A, const bf16_t* __restrict__ W,
                  float* __restrict__ C, int M, int N, int K)
{
  __shared__ bf16_t As[2][128][32];
  __shared__ bf16_t Bs[2][128][32];

  const int tid  = threadIdx.x;
  const int wave = tid >> 5;
  const int lane = tid & 31;
  const int wm   = wave >> 1;           // 0..3
  const int wn   = wave & 1;            // 0..1
  const int bm   = blockIdx.y * 128;
  const int bn   = blockIdx.x * 128;

  const int lrow = tid >> 1;            // 0..127
  const int lcol = (tid & 1) * 16;      // 0 or 16

  const bool lo  = lane < 16;
  const int  lm  = lane & 15;
  const int  akb = lo ? 0 : 8;
  const int  bkb = lo ? 0 : 8;

  v8f acc[2][4];
#pragma unroll
  for (int i = 0; i < 2; ++i)
#pragma unroll
    for (int j = 0; j < 4; ++j) acc[i][j] = v8f_zero();

  const bf16_t* ag = A + (size_t)(bm + lrow) * K + lcol;
  const bf16_t* bg = W + (size_t)(bn + lrow) * K + lcol;

  auto issue_tile = [&](int b, const bf16_t* a, const bf16_t* w) {
    async_ld_b128(&As[b][lrow][lcol],     a);
    async_ld_b128(&As[b][lrow][lcol + 8], a + 8);
    async_ld_b128(&Bs[b][lrow][lcol],     w);
    async_ld_b128(&Bs[b][lrow][lcol + 8], w + 8);
  };

  auto compute_tile = [&](int b) {
    v16bf af[2];
#pragma unroll
    for (int sm = 0; sm < 2; ++sm)
      af[sm] = load_frag_a(&As[b][wm * 32 + sm * 16 + lm][0], akb);
#pragma unroll
    for (int sn = 0; sn < 4; ++sn) {
      v16bf bfr = load_frag_b(&Bs[b][wn * 64 + sn * 16 + lm][0], bkb);
#pragma unroll
      for (int sm = 0; sm < 2; ++sm)
        acc[sm][sn] = wmma_bf16(af[sm], bfr, acc[sm][sn]);
    }
  };

  // prologue: tile 0 -> buffer 0 (4 async b128 per lane = 64 B)
  issue_tile(0, ag, bg);

  int buf = 0;
  // steady state: all tiles except the last (branch-free hot loop)
  for (int kt = 0; kt < K - 32; kt += 32) {
    ag += 32; bg += 32;
    issue_tile(buf ^ 1, ag, bg);
    __builtin_prefetch(ag + 32, 0, 3);   // two tiles ahead, keep near
    __builtin_prefetch(bg + 32, 0, 3);
    // async loads retire in order: <=4 outstanding means the older tile landed
    asm volatile("s_wait_asynccnt 0x4" ::: "memory");
    __syncthreads();
    compute_tile(buf);
    // ds reads above were consumed by WMMA (s_wait_dscnt inserted), so after
    // this barrier the just-read buffer may be overwritten by async copies.
    __syncthreads();
    buf ^= 1;
  }

  // peeled last tile
  asm volatile("s_wait_asynccnt 0x0" ::: "memory");
  __syncthreads();
  compute_tile(buf);

  // C layout: VGPR r -> M = r (lane<16) / 8+r (lane>=16), N = lane&15
#pragma unroll
  for (int sm = 0; sm < 2; ++sm)
#pragma unroll
    for (int sn = 0; sn < 4; ++sn) {
      float* cp = C + (size_t)(bm + wm * 32 + sm * 16 + (lo ? 0 : 8)) * N
                    + (bn + wn * 64 + sn * 16 + lm);
#pragma unroll
      for (int r = 0; r < 8; ++r) cp[(size_t)r * N] = acc[sm][sn][r];
    }
}

// ---------------------------------------------------------------------------
// RMSNorm (one row per block): y = norm(xin)*w [+ resid] [* scalar]
// ---------------------------------------------------------------------------
__global__ __launch_bounds__(256)
void rmsnorm_kernel(const float* __restrict__ xin, const float* __restrict__ resid,
                    const float* __restrict__ w, const float* __restrict__ scal,
                    float* __restrict__ outf, bf16_t* __restrict__ outb, int H)
{
  __shared__ float red[8];
  const int row = blockIdx.x;
  const float* xr = xin + (size_t)row * H;

  float ss = 0.0f;
  for (int i = threadIdx.x; i < H; i += blockDim.x) { float v = xr[i]; ss += v * v; }
#pragma unroll
  for (int o = 16; o > 0; o >>= 1) ss += __shfl_xor(ss, o, 32);
  if ((threadIdx.x & 31) == 0) red[threadIdx.x >> 5] = ss;
  __syncthreads();
  if (threadIdx.x < 8) {
    float v = red[threadIdx.x];
#pragma unroll
    for (int o = 4; o > 0; o >>= 1) v += __shfl_xor(v, o, 32);
    if (threadIdx.x == 0) red[0] = v;
  }
  __syncthreads();

  const float inv = rsqrtf(red[0] / (float)H + 1e-6f);
  const float s   = scal ? scal[0] : 1.0f;
  const float* rr = resid ? resid + (size_t)row * H : nullptr;
  for (int i = threadIdx.x; i < H; i += blockDim.x) {
    float y = xr[i] * inv * w[i];
    if (rr) y += rr[i];
    y *= s;
    if (outf) outf[(size_t)row * H + i] = y;
    if (outb) outb[(size_t)row * H + i] = (bf16_t)y;
  }
}

// ---------------------------------------------------------------------------
// Elementwise helpers
// ---------------------------------------------------------------------------
__global__ void f32_to_bf16_kernel(const float* __restrict__ in,
                                   bf16_t* __restrict__ out, size_t n)
{
  size_t i = (size_t)blockIdx.x * blockDim.x + threadIdx.x;
  if (i < n) out[i] = (bf16_t)in[i];
}

// v[(b*S + j)][kvh*128 + d] (f32) -> vt[(b*1024 + kvh*128 + d)][j] (bf16)
__global__ void v_to_vt_kernel(const float* __restrict__ v, bf16_t* __restrict__ vt)
{
  size_t i = (size_t)blockIdx.x * blockDim.x + threadIdx.x;
  const size_t total = (size_t)TOKENS * (N_KV * HEAD_DIM);
  if (i >= total) return;
  const int    d1  = (int)(i % (N_KV * HEAD_DIM));
  const size_t tok = i / (N_KV * HEAD_DIM);
  const int    b   = (int)(tok / SEQ);
  const int    j   = (int)(tok % SEQ);
  vt[((size_t)b * (N_KV * HEAD_DIM) + d1) * SEQ + j] = (bf16_t)v[i];
}

// fused RoPE + bf16 downconvert; one token per block, (nheads*64) threads
__global__ void rope_bf16_kernel(const float* __restrict__ xf,
                                 bf16_t* __restrict__ xb, int nheads)
{
  const int tok  = blockIdx.x;
  const int pos  = tok & (SEQ - 1);
  const int t    = threadIdx.x;
  const int head = t >> 6;
  const int i    = t & 63;
  const size_t base = (size_t)tok * ((size_t)nheads * HEAD_DIM) + (size_t)head * HEAD_DIM;
  const float x1 = xf[base + i];
  const float x2 = xf[base + 64 + i];
  const float invf = powf(10000.0f, -(float)i / 64.0f);
  const float th = (float)pos * invf;
  const float c = cosf(th), s = sinf(th);
  xb[base + i]      = (bf16_t)(x1 * c - x2 * s);
  xb[base + 64 + i] = (bf16_t)(x2 * c + x1 * s);
}

// t = gelu_exact(gate) * up  ->  bf16
__global__ void geglu_kernel(const float* __restrict__ gate,
                             const float* __restrict__ up,
                             bf16_t* __restrict__ out, size_t n)
{
  size_t i = (size_t)blockIdx.x * blockDim.x + threadIdx.x;
  if (i < n) {
    const float g  = gate[i];
    const float ge = 0.5f * g * (1.0f + erff(g * 0.70710678118654752f));
    out[i] = (bf16_t)(ge * up[i]);
  }
}

// ---------------------------------------------------------------------------
// Flash attention with sliding window. 8 waves/block, 16 q-rows per wave.
// q[B*S][2048] bf16 (post-RoPE), k[B*S][1024] bf16 (post-RoPE),
// vt[B*1024][S] bf16 (d-major), o[B*S][2048] bf16.
// ---------------------------------------------------------------------------
__global__ __launch_bounds__(256)
void attention_kernel(const bf16_t* __restrict__ q,
                      const bf16_t* __restrict__ k,
                      const bf16_t* __restrict__ vt,
                      bf16_t* __restrict__ o)
{
  __shared__ bf16_t Pl[8][16][32];   // per-wave P tile (C-layout -> A-layout)

  const int wave = threadIdx.x >> 5;
  const int lane = threadIdx.x & 31;
  const int b    = blockIdx.z;
  const int h    = blockIdx.y;
  const int kvh  = h >> 1;                 // GQA group = 2
  const int i0   = blockIdx.x * 128 + wave * 16;

  const bool lo  = lane < 16;
  const int  lm  = lane & 15;
  const int  akb = lo ? 0 : 8;
  const int  bkb = lo ? 0 : 8;

  // Q fragments for K-dim=128 (4 chunks of 32), loaded straight from global
  v16bf qf[4];
  {
    const bf16_t* qrow = q + ((size_t)(b * SEQ + i0 + lm) * (N_HEADS * HEAD_DIM))
                           + h * HEAD_DIM;
#pragma unroll
    for (int c = 0; c < 4; ++c) qf[c] = load_frag_a(qrow + c * 32, akb);
  }

  float mrow[8], lrow[8];
#pragma unroll
  for (int r = 0; r < 8; ++r) { mrow[r] = -3.0e30f; lrow[r] = 0.0f; }
  v8f acc[8];
#pragma unroll
  for (int d = 0; d < 8; ++d) acc[d] = v8f_zero();

  const float scale = 0.088388347648318447f;   // 1/sqrt(128)

  const int jhi = i0 + 15;
  int jlo = jhi - (WINDOW - 1);
  if (jlo < 0) jlo = 0;
  const int jstart = jlo & ~31;

  const bf16_t* kbase = k + ((size_t)(b * SEQ) * (N_KV * HEAD_DIM)) + kvh * HEAD_DIM;
  const bf16_t* vbase = vt + ((size_t)(b * (N_KV * HEAD_DIM) + kvh * HEAD_DIM)) * SEQ;

  for (int jt = jstart; jt <= jhi; jt += 32) {
    // ---- S = Q * K^T for two 16-key subtiles -----------------------------
    v8f sc[2];
#pragma unroll
    for (int t = 0; t < 2; ++t) {
      sc[t] = v8f_zero();
      const bf16_t* krow = kbase + (size_t)(jt + t * 16 + lm) * (N_KV * HEAD_DIM);
#pragma unroll
      for (int c = 0; c < 4; ++c) {
        v16bf kf = load_frag_b(krow + c * 32, bkb);
        sc[t] = wmma_bf16(qf[c], kf, sc[t]);
      }
    }

    // ---- causal + sliding-window mask ------------------------------------
#pragma unroll
    for (int t = 0; t < 2; ++t) {
      const int j = jt + t * 16 + lm;
#pragma unroll
      for (int r = 0; r < 8; ++r) {
        const int i = i0 + r + (lo ? 0 : 8);
        const bool ok = (j <= i) && (i - j < WINDOW);
        sc[t][r] = ok ? sc[t][r] * scale : -3.0e30f;
      }
    }

    // ---- online softmax (row reductions across the 16-lane half) ---------
#pragma unroll
    for (int r = 0; r < 8; ++r) {
      float m = fmaxf(sc[0][r], sc[1][r]);
#pragma unroll
      for (int off = 8; off > 0; off >>= 1) m = fmaxf(m, __shfl_xor(m, off, 32));
      const float nm    = fmaxf(mrow[r], m);
      const float alpha = __expf(mrow[r] - nm);
      mrow[r] = nm;
      const float p0 = (sc[0][r] < -1.0e30f) ? 0.0f : __expf(sc[0][r] - nm);
      const float p1 = (sc[1][r] < -1.0e30f) ? 0.0f : __expf(sc[1][r] - nm);
      sc[0][r] = p0; sc[1][r] = p1;
      float rs = p0 + p1;
#pragma unroll
      for (int off = 8; off > 0; off >>= 1) rs += __shfl_xor(rs, off, 32);
      lrow[r] = lrow[r] * alpha + rs;
#pragma unroll
      for (int d = 0; d < 8; ++d) acc[d][r] *= alpha;
    }

    // ---- P: C-layout f32 -> LDS -> A-layout bf16 -------------------------
    bf16_t (*pw)[32] = Pl[wave];
#pragma unroll
    for (int t = 0; t < 2; ++t)
#pragma unroll
      for (int r = 0; r < 8; ++r)
        pw[r + (lo ? 0 : 8)][t * 16 + lm] = (bf16_t)sc[t][r];
    asm volatile("s_wait_dscnt 0" ::: "memory");   // per-wave LDS RAW ordering
    v16bf pf = load_frag_a(&pw[lm][0], akb);
    asm volatile("s_wait_dscnt 0" ::: "memory");   // loads done before next stores

    // ---- O += P * V  (V is d-major so B-frags are contiguous) ------------
#pragma unroll
    for (int d = 0; d < 8; ++d) {
      const bf16_t* vrow = vbase + (size_t)(d * 16 + lm) * SEQ + jt;
      v16bf vf = load_frag_b(vrow, bkb);
      acc[d] = wmma_bf16(pf, vf, acc[d]);
    }
  }

  // ---- epilogue: divide by row sum, write bf16 ---------------------------
  bf16_t* orow = o + ((size_t)(b * SEQ + i0) * (N_HEADS * HEAD_DIM)) + h * HEAD_DIM;
#pragma unroll
  for (int d = 0; d < 8; ++d)
#pragma unroll
    for (int r = 0; r < 8; ++r) {
      const int m = r + (lo ? 0 : 8);
      orow[(size_t)m * (N_HEADS * HEAD_DIM) + d * 16 + lm] =
          (bf16_t)(acc[d][r] / lrow[r]);
    }
}

// ---------------------------------------------------------------------------
// Host orchestration
// ---------------------------------------------------------------------------
extern "C" void kernel_launch(void* const* d_in, const int* in_sizes, int n_in,
                              void* d_out, int out_size, void* d_ws, size_t ws_size,
                              hipStream_t stream)
{
  (void)in_sizes; (void)n_in; (void)out_size; (void)ws_size;
  const float* x    = (const float*)d_in[0];
  const float* wq   = (const float*)d_in[1];
  const float* wk   = (const float*)d_in[2];
  const float* wv   = (const float*)d_in[3];
  const float* wo   = (const float*)d_in[4];
  const float* w_in = (const float*)d_in[5];
  const float* w_pa = (const float*)d_in[6];
  const float* w_pf = (const float*)d_in[7];
  const float* w_po = (const float*)d_in[8];
  const float* wg   = (const float*)d_in[9];
  const float* wu   = (const float*)d_in[10];
  const float* wd   = (const float*)d_in[11];
  const float* ls   = (const float*)d_in[12];
  float* out = (float*)d_out;

  char* cur = (char*)d_ws;
  auto take = [&](size_t bytes) -> char* {
    char* p = cur;
    cur += (bytes + 255) & ~(size_t)255;
    return p;
  };

  bf16_t* h_b  = (bf16_t*)take((size_t)TOKENS * H_SIZE * 2);
  bf16_t* wq_b = (bf16_t*)take((size_t)2048 * 2048 * 2);
  bf16_t* wk_b = (bf16_t*)take((size_t)1024 * 2048 * 2);
  bf16_t* wv_b = (bf16_t*)take((size_t)1024 * 2048 * 2);
  bf16_t* wo_b = (bf16_t*)take((size_t)2048 * 2048 * 2);
  bf16_t* wg_b = (bf16_t*)take((size_t)INTER * 2048 * 2);
  bf16_t* wu_b = (bf16_t*)take((size_t)INTER * 2048 * 2);
  bf16_t* wd_b = (bf16_t*)take((size_t)2048 * INTER * 2);
  float*  q_f  = (float*) take((size_t)TOKENS * 2048 * 4);
  float*  k_f  = (float*) take((size_t)TOKENS * 1024 * 4);
  float*  v_f  = (float*) take((size_t)TOKENS * 1024 * 4);
  bf16_t* q_b  = (bf16_t*)take((size_t)TOKENS * 2048 * 2);
  bf16_t* k_b  = (bf16_t*)take((size_t)TOKENS * 1024 * 2);
  bf16_t* vt_b = (bf16_t*)take((size_t)BATCH * 1024 * SEQ * 2);
  bf16_t* at_b = (bf16_t*)take((size_t)TOKENS * 2048 * 2);
  float*  ao_f = (float*) take((size_t)TOKENS * 2048 * 4);
  float*  h2_f = (float*) take((size_t)TOKENS * 2048 * 4);
  bf16_t* fi_b = (bf16_t*)take((size_t)TOKENS * 2048 * 2);
  float*  ga_f = (float*) take((size_t)TOKENS * INTER * 4);
  float*  up_f = (float*) take((size_t)TOKENS * INTER * 4);
  bf16_t* t_b  = (bf16_t*)take((size_t)TOKENS * INTER * 2);
  float*  fo_f = (float*) take((size_t)TOKENS * 2048 * 4);

  auto cvt = [&](const float* src, bf16_t* dst, size_t n) {
    f32_to_bf16_kernel<<<dim3((unsigned)((n + 255) / 256)), dim3(256), 0, stream>>>(src, dst, n);
  };

  // one-time weight downconvert: all bf16 weights (~126 MB) fit in 192 MB L2
  cvt(wq, wq_b, (size_t)2048 * 2048);
  cvt(wk, wk_b, (size_t)1024 * 2048);
  cvt(wv, wv_b, (size_t)1024 * 2048);
  cvt(wo, wo_b, (size_t)2048 * 2048);
  cvt(wg, wg_b, (size_t)INTER * 2048);
  cvt(wu, wu_b, (size_t)INTER * 2048);
  cvt(wd, wd_b, (size_t)2048 * INTER);

  const dim3 blk(256);

  // h = rms_norm(x, w_in) -> bf16
  rmsnorm_kernel<<<TOKENS, blk, 0, stream>>>(x, nullptr, w_in, nullptr, nullptr, h_b, H_SIZE);

  // QKV projections
  gemm_bf16_nt<<<dim3(2048 / 128, TOKENS / 128), blk, 0, stream>>>(h_b, wq_b, q_f, TOKENS, 2048, 2048);
  gemm_bf16_nt<<<dim3(1024 / 128, TOKENS / 128), blk, 0, stream>>>(h_b, wk_b, k_f, TOKENS, 1024, 2048);
  gemm_bf16_nt<<<dim3(1024 / 128, TOKENS / 128), blk, 0, stream>>>(h_b, wv_b, v_f, TOKENS, 1024, 2048);

  // RoPE + bf16; V transposed to d-major
  rope_bf16_kernel<<<TOKENS, N_HEADS * 64, 0, stream>>>(q_f, q_b, N_HEADS);
  rope_bf16_kernel<<<TOKENS, N_KV * 64, 0, stream>>>(k_f, k_b, N_KV);
  v_to_vt_kernel<<<dim3((unsigned)(((size_t)TOKENS * 1024 + 255) / 256)), blk, 0, stream>>>(v_f, vt_b);

  // sliding-window flash attention
  attention_kernel<<<dim3(SEQ / 128, N_HEADS, BATCH), blk, 0, stream>>>(q_b, k_b, vt_b, at_b);

  // output projection + residual/norms
  gemm_bf16_nt<<<dim3(2048 / 128, TOKENS / 128), blk, 0, stream>>>(at_b, wo_b, ao_f, TOKENS, 2048, 2048);
  rmsnorm_kernel<<<TOKENS, blk, 0, stream>>>(ao_f, x, w_pa, nullptr, h2_f, nullptr, H_SIZE);
  rmsnorm_kernel<<<TOKENS, blk, 0, stream>>>(h2_f, nullptr, w_pf, nullptr, nullptr, fi_b, H_SIZE);

  // FFN
  gemm_bf16_nt<<<dim3(INTER / 128, TOKENS / 128), blk, 0, stream>>>(fi_b, wg_b, ga_f, TOKENS, INTER, 2048);
  gemm_bf16_nt<<<dim3(INTER / 128, TOKENS / 128), blk, 0, stream>>>(fi_b, wu_b, up_f, TOKENS, INTER, 2048);
  geglu_kernel<<<dim3((unsigned)(((size_t)TOKENS * INTER + 255) / 256)), blk, 0, stream>>>(
      ga_f, up_f, t_b, (size_t)TOKENS * INTER);
  gemm_bf16_nt<<<dim3(2048 / 128, TOKENS / 128), blk, 0, stream>>>(t_b, wd_b, fo_f, TOKENS, 2048, 8192);

  // out = (h2 + rms_norm(ffn_out, w_po)) * layer_scalar
  rmsnorm_kernel<<<TOKENS, blk, 0, stream>>>(fo_f, h2_f, w_po, ls, out, nullptr, H_SIZE);
}
